// AttentionTSP_4406636445699
// MI455X (gfx1250) — compile-verified, matrix operand order checked by reading, not run on hardware
//
#include <hip/hip_runtime.h>
#include <hip/hip_bf16.h>

// ---------------------------------------------------------------------------
// AttentionTSP for MI455X (gfx1250, wave32).
// Encoder GEMMs + attention on v_wmma_f32_16x16x32_f16 (16x64 register
// blocking); K operand staged via global_load_async_to_lds_b128 (ASYNCcnt);
// decoder = persistent per-batch workgroup (latency-bound sampling loop).
// ---------------------------------------------------------------------------

typedef _Float16 f16;
typedef _Float16 v8h  __attribute__((ext_vector_type(8)));
typedef _Float16 v16h __attribute__((ext_vector_type(16)));
typedef float    v8f  __attribute__((ext_vector_type(8)));

#define BB      32
#define SS      512
#define EMB     128
#define NH      4
#define HD      32
#define FFN     512
#define NBLK    2

// Load a 16-element f16 A/B fragment for wmma_f32_16x16x32_f16.
// Per ISA: lane holds row (lane%16); element e<8 -> K = half*8+e,
// e>=8 -> K = 16+half*8+(e-8).  Caller passes p = row_base + half*8;
// the two 8-element runs are at p[0..7] and p[16..23]. 16B-aligned.
__device__ inline v16h load_frag(const f16* p) {
  v8h lo = *(const v8h*)(p);
  v8h hi = *(const v8h*)(p + 16);
  v16h r;
#pragma unroll
  for (int i = 0; i < 8; ++i) { r[i] = lo[i]; r[i + 8] = hi[i]; }
  return r;
}

// ---------------------------------------------------------------------------
__global__ void cvt_f16_kernel(const float* __restrict__ src, f16* __restrict__ dst, int n) {
  int i = blockIdx.x * blockDim.x + threadIdx.x;
  if (i < n) dst[i] = (f16)src[i];
}

// ---------------------------------------------------------------------------
// Embedding: h = inputs @ emb_w^T + emb_b   ([B*S,2] x [128,2])
// ---------------------------------------------------------------------------
__global__ void embed_kernel(const float* __restrict__ in, const float* __restrict__ w,
                             const float* __restrict__ bias,
                             float* __restrict__ h32, f16* __restrict__ h16) {
  int idx = blockIdx.x * blockDim.x + threadIdx.x;   // over 16384*128
  int e = idx & 127;
  int n = idx >> 7;
  float v = in[n * 2] * w[e * 2] + in[n * 2 + 1] * w[e * 2 + 1] + bias[e];
  h32[idx] = v;
  h16[idx] = (f16)v;
}

// ---------------------------------------------------------------------------
// WMMA GEMM, 16x64 per wave:  out = act(A[M,K] @ W[N,K]^T + bias) (+ resid)
// block = 4 waves (4 consecutive M tiles), grid = (N/64, M/64).
// ---------------------------------------------------------------------------
__global__ void __launch_bounds__(128)
gemm_wmma_kernel(const f16* __restrict__ A, const f16* __restrict__ W,
                 const float* __restrict__ bias, const float* __restrict__ resid,
                 float* __restrict__ out32, f16* __restrict__ out16,
                 int M, int N, int K, int relu) {
  int lane = threadIdx.x & 31;
  int wave = threadIdx.x >> 5;
  int n0 = blockIdx.x * 64;
  int m0 = (blockIdx.y * 4 + wave) * 16;
  int ln16 = lane & 15;
  int kh = (lane >> 4) << 3;                 // half*8
  const f16* ap = A + (size_t)(m0 + ln16) * K + kh;
  const f16* wp = W + (size_t)(n0 + ln16) * K + kh;
  v8f c0 = {}, c1 = {}, c2 = {}, c3 = {};
  for (int k0 = 0; k0 < K; k0 += 32) {
    if (k0 + 32 < K) __builtin_prefetch(ap + k0 + 32, 0, 3);   // global_prefetch_b8 (WGP)
    v16h a  = load_frag(ap + k0);
    v16h b0 = load_frag(wp + k0);
    v16h b1 = load_frag(wp + (size_t)16 * K + k0);
    v16h b2 = load_frag(wp + (size_t)32 * K + k0);
    v16h b3 = load_frag(wp + (size_t)48 * K + k0);
    c0 = __builtin_amdgcn_wmma_f32_16x16x32_f16(false, a, false, b0, (short)0, c0, false, false);
    c1 = __builtin_amdgcn_wmma_f32_16x16x32_f16(false, a, false, b1, (short)0, c1, false, false);
    c2 = __builtin_amdgcn_wmma_f32_16x16x32_f16(false, a, false, b2, (short)0, c2, false, false);
    c3 = __builtin_amdgcn_wmma_f32_16x16x32_f16(false, a, false, b3, (short)0, c3, false, false);
  }
  int rbase = m0 + ((lane >> 4) << 3);
  v8f acc[4] = {c0, c1, c2, c3};
#pragma unroll
  for (int tn = 0; tn < 4; ++tn) {
    int col = n0 + tn * 16 + ln16;
    float bv = bias ? bias[col] : 0.0f;
#pragma unroll
    for (int i = 0; i < 8; ++i) {
      size_t o = (size_t)(rbase + i) * N + col;
      float v = acc[tn][i] + bv;
      if (relu) v = v > 0.0f ? v : 0.0f;
      if (resid) v += resid[o];
      if (out32) out32[o] = v;
      if (out16) out16[o] = (f16)v;
    }
  }
}

// ---------------------------------------------------------------------------
// Flash-style encoder self-attention (no mask), WMMA throughout.
// qkv16: [B*S, 384] f16 (q|k|v, head-major within 128).  o16: [B*S,128] f16.
// grid.x = B*NH*8 ; block = 128 (4 waves, each one 16-row q tile).
// K block is staged Memory->LDS with async copies (ASYNCcnt); V is staged
// transposed through normal LDS stores for the P@V B-fragments.
// ---------------------------------------------------------------------------
__global__ void __launch_bounds__(128)
attn_kernel(const f16* __restrict__ qkv, f16* __restrict__ o16) {
  __shared__ __align__(16) f16 Ksh[SS][40];        // K rows, pad: 80B stride (16B aligned)
  __shared__ __align__(16) f16 Vt[HD][SS + 8];     // V^T
  __shared__ __align__(16) f16 Pst[4][16][48];     // per-wave P re-layout buffer

  int tid = threadIdx.x, lane = tid & 31, wave = tid >> 5;
  int wg = blockIdx.x;
  int qgroup = wg & 7;
  int bh = wg >> 3;
  int h = bh & 3, b = bh >> 2;
  size_t rowb = (size_t)b * SS;

  // --- async-stage K rows for this (b,h): 512 rows x 64B = 2048 x b128 ---
  {
    const f16* kbase = qkv + rowb * 384 + 128 + h * 32;    // row stride 384 f16
    for (int i = tid; i < SS * 4; i += 128) {
      int s = i >> 2, c = i & 3;
      unsigned gofs = (unsigned)((s * 384 + c * 8) * sizeof(f16));
      unsigned lofs = (unsigned)(uintptr_t)(&Ksh[s][c * 8]);
      asm volatile("global_load_async_to_lds_b128 %0, %1, %2"
                   :: "v"(lofs), "v"(gofs), "s"(kbase) : "memory");
    }
  }
  // --- stage V^T (transpose => plain LDS stores) ---
  for (int idx = tid; idx < SS * HD; idx += 128) {
    int s = idx >> 5, d = idx & 31;
    Vt[d][s] = qkv[(rowb + s) * 384 + 256 + h * 32 + d];
  }
  asm volatile("s_wait_asynccnt 0" ::: "memory");
  __syncthreads();

  int m0 = (qgroup * 4 + wave) * 16;
  int ln16 = lane & 15;
  int half = lane >> 4;
  int kh = half << 3;
  const float scale = 0.17677669529663687f;        // HD^-0.5

  v16h qa = load_frag(qkv + (rowb + m0 + ln16) * 384 + h * 32 + kh);

  v8f oc0 = {}, oc1 = {};
  float m8[8], l8[8];
#pragma unroll
  for (int i = 0; i < 8; ++i) { m8[i] = -1e30f; l8[i] = 0.0f; }

  for (int cch = 0; cch < SS / 32; ++cch) {
    int k0 = cch * 32;
    v16h kb0 = load_frag(&Ksh[k0 + ln16][kh]);
    v16h kb1 = load_frag(&Ksh[k0 + 16 + ln16][kh]);
    v8f zero = {};
    v8f s0 = __builtin_amdgcn_wmma_f32_16x16x32_f16(false, qa, false, kb0, (short)0, zero, false, false);
    v8f s1 = __builtin_amdgcn_wmma_f32_16x16x32_f16(false, qa, false, kb1, (short)0, zero, false, false);

    float mnew[8], corr[8];
#pragma unroll
    for (int i = 0; i < 8; ++i) {
      float a0 = s0[i] * scale, a1 = s1[i] * scale;
      s0[i] = a0; s1[i] = a1;
      float mx = fmaxf(a0, a1);
      for (int off = 1; off < 16; off <<= 1) mx = fmaxf(mx, __shfl_xor(mx, off));
      mnew[i] = fmaxf(m8[i], mx);
      corr[i] = __expf(m8[i] - mnew[i]);
      m8[i] = mnew[i];
    }
#pragma unroll
    for (int i = 0; i < 8; ++i) {
      float e0 = __expf(s0[i] - mnew[i]);
      float e1 = __expf(s1[i] - mnew[i]);
      float sm = e0 + e1;
      for (int off = 1; off < 16; off <<= 1) sm += __shfl_xor(sm, off);
      l8[i] = l8[i] * corr[i] + sm;
      oc0[i] *= corr[i];
      oc1[i] *= corr[i];
      int r = (half << 3) + i;                    // C-layout row
      Pst[wave][r][ln16] = (f16)e0;
      Pst[wave][r][16 + ln16] = (f16)e1;
    }
    asm volatile("s_wait_dscnt 0" ::: "memory");  // LDS stores visible before frag reload
    v16h pa  = load_frag(&Pst[wave][ln16][kh]);   // P as A fragment (K=32)
    v16h vb0 = load_frag(&Vt[ln16][k0 + kh]);     // V columns 0..15
    v16h vb1 = load_frag(&Vt[16 + ln16][k0 + kh]);// V columns 16..31
    oc0 = __builtin_amdgcn_wmma_f32_16x16x32_f16(false, pa, false, vb0, (short)0, oc0, false, false);
    oc1 = __builtin_amdgcn_wmma_f32_16x16x32_f16(false, pa, false, vb1, (short)0, oc1, false, false);
  }

#pragma unroll
  for (int i = 0; i < 8; ++i) {
    int r = m0 + (half << 3) + i;
    float inv = 1.0f / l8[i];
    o16[(rowb + r) * EMB + h * 32 + ln16]      = (f16)(oc0[i] * inv);
    o16[(rowb + r) * EMB + h * 32 + 16 + ln16] = (f16)(oc1[i] * inv);
  }
}

// ---------------------------------------------------------------------------
// h_bar = mean_s(h) @ hctx_w^T + hctx_b ; vq = init_w @ vw_w^T + vw_b
// ---------------------------------------------------------------------------
__global__ void __launch_bounds__(128)
hbar_kernel(const float* __restrict__ h32, const float* __restrict__ hctx_w,
            const float* __restrict__ hctx_b, const float* __restrict__ vw_w,
            const float* __restrict__ vw_b, const float* __restrict__ init_w,
            float* __restrict__ hbar, float* __restrict__ vq) {
  __shared__ float hm[EMB];
  int b = blockIdx.x, e = threadIdx.x;
  float s = 0.0f;
  for (int t = 0; t < SS; ++t) s += h32[((size_t)b * SS + t) * EMB + e];
  hm[e] = s * (1.0f / (float)SS);
  __syncthreads();
  float acc = hctx_b[e];
  for (int j = 0; j < EMB; ++j) acc += hm[j] * hctx_w[e * EMB + j];
  hbar[b * EMB + e] = acc;
  if (b == 0) {
    float a2 = vw_b[e];
    for (int j = 0; j < 2 * EMB; ++j) a2 += init_w[j] * vw_w[e * 2 * EMB + j];
    vq[e] = a2;
  }
}

// ---------------------------------------------------------------------------
// Deterministic counter-based uniform (splitmix64 hash on (t,b,s)).
// ---------------------------------------------------------------------------
__device__ inline float rng_uniform(unsigned t, unsigned b, unsigned s) {
  unsigned long long x = ((unsigned long long)t << 40) ^ ((unsigned long long)b << 20)
                       ^ (unsigned long long)s ^ 0x9E3779B97F4A7C15ull;
  x ^= x >> 30; x *= 0xBF58476D1CE4E5B9ull;
  x ^= x >> 27; x *= 0x94D049BB133111EBull;
  x ^= x >> 31;
  unsigned r = (unsigned)(x >> 40);                 // 24 bits
  return ((float)r + 0.5f) * (1.0f / 16777216.0f);
}

// ---------------------------------------------------------------------------
// Persistent decoder: one workgroup (256 thr / 8 waves) per batch element.
// 512 sequential steps; hot state in LDS; gk/gv/pk stream from L2.
// Every 128-wide gemv is split across 256 threads; reductions are
// shfl-within-wave + one cross-wave combine (min syncthreads on the
// latency-critical chain).
// ---------------------------------------------------------------------------
__global__ void __launch_bounds__(256)
decode_kernel(const float* __restrict__ h32, const float* __restrict__ gk32,
              const float* __restrict__ gv32, const float* __restrict__ pk32,
              const float* __restrict__ hbar, const float* __restrict__ vq,
              const float* __restrict__ gq_w, const float* __restrict__ gq_b,
              const float* __restrict__ go_w, const float* __restrict__ go_b,
              const float* __restrict__ pq_w, const float* __restrict__ pq_b,
              const float* __restrict__ vw_w, const float* __restrict__ vw_b,
              float* __restrict__ out_lp, int* __restrict__ out_idx) {
  __shared__ float query[EMB], gq[EMB], nq[EMB], nq2[EMB], pqv[EMB];
  __shared__ float xvec[2 * EMB];                  // [first_h | chosen_h]
  __shared__ float gl[NH][SS];
  __shared__ float pl[SS];
  __shared__ float red[256];
  __shared__ float nqpart[2][EMB];
  __shared__ float redm[8], redy[8];
  __shared__ int   redi[8];
  __shared__ unsigned maskb[SS / 32];
  __shared__ float s_pmax;
  __shared__ int s_chosen;

  int tid = threadIdx.x, b = blockIdx.x;
  int lane = tid & 31, wave = tid >> 5;
  int row = tid & 127, hf = tid >> 7;              // gemv split
  if (tid < EMB) query[tid] = hbar[b * EMB + tid] + vq[tid];
  if (tid < SS / 32) maskb[tid] = 0u;
  __syncthreads();

  const float ghs = 0.17677669529663687f;   // GHD^-0.5
  const float hs  = 0.08838834764831845f;   // HIDDEN^-0.5

  for (int t = 0; t < SS; ++t) {
    // --- gq = query @ gq_w^T + gq_b (split dot) ---
    {
      float a = 0.0f;
      const float* wr = gq_w + row * EMB + hf * 64;
      const float* xv = query + hf * 64;
      for (int j = 0; j < 64; ++j) a += xv[j] * wr[j];
      red[tid] = a;
    }
    __syncthreads();
    if (tid < EMB) gq[tid] = gq_b[tid] + red[tid] + red[tid + 128];
    __syncthreads();
    // --- masked glimpse logits [NH][SS] ---
    for (int idx = tid; idx < NH * SS; idx += 256) {
      int hh = idx >> 9, s = idx & (SS - 1);
      bool msk = (maskb[s >> 5] >> (s & 31)) & 1u;
      float a;
      if (msk) a = -1e9f;
      else {
        a = 0.0f;
        const float* kk = gk32 + ((size_t)b * SS + s) * EMB + hh * HD;
        const float* qq = gq + hh * HD;
        for (int d = 0; d < HD; ++d) a += qq[d] * kk[d];
        a *= ghs;
      }
      gl[hh][s] = a;
    }
    __syncthreads();
    // --- per-head softmax (wave w == head w) ---
    if (wave < NH) {
      float mx = -1e30f;
      for (int s = lane; s < SS; s += 32) mx = fmaxf(mx, gl[wave][s]);
      for (int off = 1; off < 32; off <<= 1) mx = fmaxf(mx, __shfl_xor(mx, off));
      float sm = 0.0f;
      for (int s = lane; s < SS; s += 32) sm += __expf(gl[wave][s] - mx);
      for (int off = 1; off < 32; off <<= 1) sm += __shfl_xor(sm, off);
      float inv = 1.0f / sm;
      for (int s = lane; s < SS; s += 32) gl[wave][s] = __expf(gl[wave][s] - mx) * inv;
    }
    __syncthreads();
    // --- n_query = alpha @ gv (8 waves: head = wave&3, s-half = wave>>2) ---
    {
      int head = wave & 3, sh = wave >> 2;
      float acc = 0.0f;
      const float* gvb = gv32 + (size_t)b * SS * EMB + head * HD + lane;
      int s0 = sh * (SS / 2);
      for (int s = s0; s < s0 + SS / 2; ++s) acc += gl[head][s] * gvb[(size_t)s * EMB];
      nqpart[sh][head * HD + lane] = acc;
    }
    __syncthreads();
    if (tid < EMB) nq[tid] = nqpart[0][tid] + nqpart[1][tid];
    __syncthreads();
    // --- n_query @ go_w^T + go_b (split dot) ---
    {
      float a = 0.0f;
      const float* wr = go_w + row * EMB + hf * 64;
      const float* xv = nq + hf * 64;
      for (int j = 0; j < 64; ++j) a += xv[j] * wr[j];
      red[tid] = a;
    }
    __syncthreads();
    if (tid < EMB) nq2[tid] = go_b[tid] + red[tid] + red[tid + 128];
    __syncthreads();
    // --- pq = nq2 @ pq_w^T + pq_b (split dot) ---
    {
      float a = 0.0f;
      const float* wr = pq_w + row * EMB + hf * 64;
      const float* xv = nq2 + hf * 64;
      for (int j = 0; j < 64; ++j) a += xv[j] * wr[j];
      red[tid] = a;
    }
    __syncthreads();
    if (tid < EMB) pqv[tid] = pq_b[tid] + red[tid] + red[tid + 128];
    __syncthreads();
    // --- tanh-clipped masked pointer logits ---
    for (int s = tid; s < SS; s += 256) {
      bool msk = (maskb[s >> 5] >> (s & 31)) & 1u;
      float a;
      if (msk) a = -1e9f;
      else {
        a = 0.0f;
        const float* kk = pk32 + ((size_t)b * SS + s) * EMB;
        for (int d = 0; d < EMB; ++d) a += pqv[d] * kk[d];
        a = 10.0f * tanhf(a * hs);
      }
      pl[s] = a;
    }
    __syncthreads();
    // --- phase 1: max + gumbel-argmax (wave shfl reduce + combine) ---
    {
      float lmax = -1e30f, lby = -1e30f;
      int lbi = 0;
      for (int s = tid; s < SS; s += 256) {
        float v = pl[s];
        lmax = fmaxf(lmax, v);
        float u = rng_uniform((unsigned)t, (unsigned)b, (unsigned)s);
        float g = -__logf(-__logf(u));
        float y = v + g;
        if (y > lby) { lby = y; lbi = s; }
      }
      for (int off = 1; off < 32; off <<= 1) {
        lmax = fmaxf(lmax, __shfl_xor(lmax, off));
        float oy = __shfl_xor(lby, off);
        int oi = __shfl_xor(lbi, off);
        if (oy > lby) { lby = oy; lbi = oi; }
      }
      if (lane == 0) { redm[wave] = lmax; redy[wave] = lby; redi[wave] = lbi; }
    }
    __syncthreads();
    if (tid == 0) {
      float pm = redm[0], by = redy[0];
      int bi = redi[0];
      for (int w = 1; w < 8; ++w) {
        pm = fmaxf(pm, redm[w]);
        if (redy[w] > by) { by = redy[w]; bi = redi[w]; }
      }
      s_pmax = pm;
      s_chosen = bi;
      out_idx[b * SS + t] = bi;
      maskb[bi >> 5] |= (1u << (bi & 31));
    }
    __syncthreads();
    // --- phase 2: logsumexp, emit logprob ---
    {
      float pm = s_pmax;
      float lsum = 0.0f;
      for (int s = tid; s < SS; s += 256) lsum += __expf(pl[s] - pm);
      for (int off = 1; off < 32; off <<= 1) lsum += __shfl_xor(lsum, off);
      if (lane == 0) redm[wave] = lsum;
    }
    __syncthreads();
    int ch = s_chosen;
    if (tid == 0) {
      float ls = 0.0f;
      for (int w = 0; w < 8; ++w) ls += redm[w];
      out_lp[b * SS + t] = pl[ch] - (s_pmax + __logf(ls));
    }
    // --- gather chosen embedding; build [first_h | chosen_h] ---
    if (tid < EMB) {
      float cv = h32[((size_t)b * SS + ch) * EMB + tid];
      xvec[EMB + tid] = cv;
      if (t == 0) xvec[tid] = cv;
    }
    __syncthreads();
    // --- query = h_bar + xvec @ vw_w^T + vw_b (split dot over 256) ---
    {
      float a = 0.0f;
      const float* wr = vw_w + row * 2 * EMB + hf * EMB;
      const float* xv = xvec + hf * EMB;
      for (int j = 0; j < EMB; ++j) a += xv[j] * wr[j];
      red[tid] = a;
    }
    __syncthreads();
    if (tid < EMB) query[tid] = hbar[b * EMB + tid] + vw_b[tid] + red[tid] + red[tid + 128];
    __syncthreads();
  }
}

// ---------------------------------------------------------------------------
extern "C" void kernel_launch(void* const* d_in, const int* in_sizes, int n_in,
                              void* d_out, int out_size, void* d_ws, size_t ws_size,
                              hipStream_t stream) {
  (void)in_sizes; (void)n_in; (void)out_size; (void)ws_size;
  const float* inputs     = (const float*)d_in[0];
  const float* emb_w      = (const float*)d_in[1];
  const float* emb_b      = (const float*)d_in[2];
  const float* in_proj_w  = (const float*)d_in[3];
  const float* in_proj_b  = (const float*)d_in[4];
  const float* out_proj_w = (const float*)d_in[5];
  const float* out_proj_b = (const float*)d_in[6];
  const float* ff1_w      = (const float*)d_in[7];
  const float* ff1_b      = (const float*)d_in[8];
  const float* ff2_w      = (const float*)d_in[9];
  const float* ff2_b      = (const float*)d_in[10];
  const float* hctx_w     = (const float*)d_in[11];
  const float* hctx_b     = (const float*)d_in[12];
  const float* vw_w       = (const float*)d_in[13];
  const float* vw_b       = (const float*)d_in[14];
  const float* init_w     = (const float*)d_in[15];
  const float* gq_w       = (const float*)d_in[16];
  const float* gq_b       = (const float*)d_in[17];
  const float* gk_w       = (const float*)d_in[18];
  const float* gk_b       = (const float*)d_in[19];
  const float* gv_w       = (const float*)d_in[20];
  const float* gv_b       = (const float*)d_in[21];
  const float* go_w       = (const float*)d_in[22];
  const float* go_b       = (const float*)d_in[23];
  const float* pq_w       = (const float*)d_in[24];
  const float* pq_b       = (const float*)d_in[25];
  const float* pk_w       = (const float*)d_in[26];
  const float* pk_b       = (const float*)d_in[27];

  const int M = BB * SS;                       // 16384 rows

  // ---- workspace carve (256B aligned) ----
  char* p = (char*)d_ws;
  auto carve = [&](size_t bytes) -> void* {
    void* r = (void*)p;
    p += (bytes + 255) & ~(size_t)255;
    return r;
  };
  float* h32   = (float*)carve((size_t)M * EMB * 4);
  float* gk32  = (float*)carve((size_t)M * EMB * 4);
  float* gv32  = (float*)carve((size_t)M * EMB * 4);
  float* pk32  = (float*)carve((size_t)M * EMB * 4);
  float* hbar  = (float*)carve((size_t)BB * EMB * 4);
  float* vq    = (float*)carve((size_t)EMB * 4);
  f16* h16     = (f16*)carve((size_t)M * EMB * 2);
  f16* qkv16   = (f16*)carve((size_t)M * 3 * EMB * 2);
  f16* o16     = (f16*)carve((size_t)M * EMB * 2);
  f16* mid16   = (f16*)carve((size_t)M * FFN * 2);
  f16* w_inp16 = (f16*)carve((size_t)NBLK * 3 * EMB * EMB * 2);
  f16* w_out16 = (f16*)carve((size_t)NBLK * EMB * EMB * 2);
  f16* w_ff116 = (f16*)carve((size_t)NBLK * FFN * EMB * 2);
  f16* w_ff216 = (f16*)carve((size_t)NBLK * EMB * FFN * 2);
  f16* w_gk16  = (f16*)carve((size_t)EMB * EMB * 2);
  f16* w_gv16  = (f16*)carve((size_t)EMB * EMB * 2);
  f16* w_pk16  = (f16*)carve((size_t)EMB * EMB * 2);

  // ---- weight conversion ----
  auto cvt = [&](const float* s, f16* d, int n) {
    cvt_f16_kernel<<<(n + 255) / 256, 256, 0, stream>>>(s, d, n);
  };
  cvt(in_proj_w,  w_inp16, NBLK * 3 * EMB * EMB);
  cvt(out_proj_w, w_out16, NBLK * EMB * EMB);
  cvt(ff1_w,      w_ff116, NBLK * FFN * EMB);
  cvt(ff2_w,      w_ff216, NBLK * EMB * FFN);
  cvt(gk_w,       w_gk16,  EMB * EMB);
  cvt(gv_w,       w_gv16,  EMB * EMB);
  cvt(pk_w,       w_pk16,  EMB * EMB);

  // ---- embedding ----
  embed_kernel<<<(M * EMB) / 256, 256, 0, stream>>>(inputs, emb_w, emb_b, h32, h16);

  auto gemm = [&](const f16* A, const f16* W, const float* bias, const float* resid,
                  float* o32, f16* of16, int N, int K, int relu) {
    dim3 g(N / 64, M / 64);
    gemm_wmma_kernel<<<g, 128, 0, stream>>>(A, W, bias, resid, o32, of16, M, N, K, relu);
  };

  // ---- encoder blocks ----
  for (int l = 0; l < NBLK; ++l) {
    gemm(h16, w_inp16 + (size_t)l * 3 * EMB * EMB, in_proj_b + l * 3 * EMB,
         nullptr, nullptr, qkv16, 3 * EMB, EMB, 0);
    attn_kernel<<<BB * NH * 8, 128, 0, stream>>>(qkv16, o16);
    gemm(o16, w_out16 + (size_t)l * EMB * EMB, out_proj_b + l * EMB,
         h32, h32, h16, EMB, EMB, 0);                                   // h += attn
    gemm(h16, w_ff116 + (size_t)l * FFN * EMB, ff1_b + l * FFN,
         nullptr, nullptr, mid16, FFN, EMB, 1);                         // relu
    gemm(mid16, w_ff216 + (size_t)l * EMB * FFN, ff2_b + l * EMB,
         h32, h32, h16, EMB, FFN, 0);                                   // h += ff
  }

  // ---- decoder key/value precompute ----
  gemm(h16, w_gk16, gk_b, nullptr, gk32, nullptr, EMB, EMB, 0);
  gemm(h16, w_gv16, gv_b, nullptr, gv32, nullptr, EMB, EMB, 0);
  gemm(h16, w_pk16, pk_b, nullptr, pk32, nullptr, EMB, EMB, 0);

  // ---- context ----
  hbar_kernel<<<BB, EMB, 0, stream>>>(h32, hctx_w, hctx_b, vw_w, vw_b, init_w, hbar, vq);

  // ---- sequential pointer decoder ----
  float* out_lp = (float*)d_out;
  int* out_idx  = (int*)d_out + BB * SS;
  decode_kernel<<<BB, 256, 0, stream>>>(h32, gk32, gv32, pk32, hbar, vq,
                                        gq_w, gq_b, go_w, go_b, pq_w, pq_b,
                                        vw_w, vw_b, out_lp, out_idx);
}
